// SimpleRGCN_31344671326736
// MI455X (gfx1250) — matrix operand order repaired and checked by reading, not compiled, not run on hardware
//
#include <hip/hip_runtime.h>
#include <hip/hip_bf16.h>

// ---------------------------------------------------------------------------
// RGCN (PyG RGCNConv, aggr='mean'), 2 layers, N=50k, R=8, E=800k, d=128.
// Bandwidth-bound (~2.5 GB HBM traffic vs ~15 GFLOP) -> GEMMs are free; run
// them on the matrix pipe (v_wmma_f32_16x16x32_bf16). Weights are pre-swizzled
// once into bf16 WMMA-fragment layout so the hot loop loads a whole B fragment
// with one 32B vector load instead of 16 strided dwords.
// ---------------------------------------------------------------------------

#define RGCN_R 8
#define RGCN_D 128

typedef __attribute__((ext_vector_type(16))) __bf16 v16bf;
typedef __attribute__((ext_vector_type(8)))  float  v8f;

// ---------------------------------------------------------------- zero fill
__global__ void k_zero_f4(float* __restrict__ p, size_t nfloats) {
  size_t i = ((size_t)blockIdx.x * blockDim.x + threadIdx.x) * 4;
  size_t stride = (size_t)gridDim.x * blockDim.x * 4;
  float4 z = make_float4(0.f, 0.f, 0.f, 0.f);
  for (; i + 3 < nfloats; i += stride) *(float4*)(p + i) = z;
}

// ---------------------------------------------------- weight pre-swizzle
// Pack W (8 relations + root, each 128x128 f32, row-major [k][o]) into bf16
// WMMA B-fragment layout: frag index (rr, kt, ct), per-lane 16 bf16 where
// lanes 0-15 hold Klocal=0..15 (col = ct*16 + lane), lanes 16-31 Klocal=16..31.
// 9*4*8 fragments * 32 lanes * 32B = 288KB, computed once, L2-resident.
__global__ __launch_bounds__(256) void k_prepack_w(
    const float* __restrict__ Wrel, const float* __restrict__ Wroot,
    unsigned short* __restrict__ Bpk) {
  int t = blockIdx.x * blockDim.x + threadIdx.x;  // one thread per (frag,lane)
  if (t >= 9 * 4 * 8 * 32) return;
  int lane = t & 31;
  int ct = (t >> 5) & 7;
  int kt = (t >> 8) & 3;
  int rr = t >> 10;  // 0..7 relations, 8 = root
  const float* W = (rr < RGCN_R) ? Wrel + (size_t)rr * RGCN_D * RGCN_D : Wroot;
  int half = lane >> 4, n = lane & 15;
  union { unsigned short u[16]; uint4 q[2]; } buf;
#pragma unroll
  for (int e = 0; e < 16; ++e) {
    int K = kt * 32 + half * 16 + e;
    __bf16 b = (__bf16)W[(size_t)K * RGCN_D + ct * 16 + n];
    buf.u[e] = __builtin_bit_cast(unsigned short, b);
  }
  uint4* dst = (uint4*)Bpk + (size_t)t * 2;
  dst[0] = buf.q[0];
  dst[1] = buf.q[1];
}

// ------------------------------------------------------------- edge scatter
// One 32-lane wave per edge: 512B coalesced float4 read of feat[src], then
// 4x global_atomic_add_f32 into the (dst, etype) aggregate row.
__global__ __launch_bounds__(256) void k_scatter(
    const float* __restrict__ feat, const int* __restrict__ src,
    const int* __restrict__ dst, const int* __restrict__ et,
    float* __restrict__ agg, float* __restrict__ cnt, int E, int addCount) {
  int gt = blockIdx.x * blockDim.x + threadIdx.x;
  int e = gt >> 5;
  int lane = gt & 31;
  if (e >= E) return;
  int s = src[e], d = dst[e], t = et[e];
  const float* xr = feat + (size_t)s * RGCN_D;
  float* ar = agg + ((size_t)d * RGCN_R + t) * RGCN_D;
  int k = lane * 4;
  float4 v = *(const float4*)(xr + k);
  atomicAdd(ar + k + 0, v.x);
  atomicAdd(ar + k + 1, v.y);
  atomicAdd(ar + k + 2, v.z);
  atomicAdd(ar + k + 3, v.w);
  if (addCount && lane == 0) atomicAdd(&cnt[(size_t)d * RGCN_R + t], 1.0f);
}

// --------------------------------------------------------- layer 1 (WMMA)
// Each wave owns one 16-node tile and ALL 8 output col tiles (8x v8f accs):
// the A fragment is built exactly once per (operand, K-chunk) and feeds 8
// back-to-back WMMAs against prepacked B fragments (one 32B load each).
// After consuming its agg rows the wave re-zeros them (no barrier needed:
// sole reader) so the layer-2 scatter can reuse the buffer.
__global__ __launch_bounds__(128) void k_layer1_wmma(
    float* __restrict__ agg, const float* __restrict__ cnt,
    const float* __restrict__ x, const unsigned short* __restrict__ Bpk,
    const float* __restrict__ bias, float* __restrict__ h, int nNodes) {
  const int lane = threadIdx.x & 31;
  const int wave = threadIdx.x >> 5;
  const int tile0 = blockIdx.x * 64 + wave * 16;  // this wave's 16 nodes
  const int m = lane & 15;
  const int half = lane >> 4;
  const int ncol = lane & 15;
  int node = tile0 + m;
  if (node >= nNodes) node = nNodes - 1;  // clamp loads; stores are guarded

  const v16bf* Bfrag = (const v16bf*)Bpk;
  v8f acc[8];
#pragma unroll
  for (int ct = 0; ct < 8; ++ct) acc[ct] = (v8f){};

  // One operand: A row = arow[0..127] * s against weight-set rr.
  auto mac_operand = [&](const float* arow, float s, int rr) {
#pragma unroll
    for (int kt = 0; kt < 4; ++kt) {
      // A fragment: contiguous K order matching prepacked B.
      // lanes 0-15: Klocal 0..15
      // lanes16-31: Klocal 16..31
      int kb = kt * 32 + half * 16;
      float4 a0 = *(const float4*)(arow + kb);
      float4 a1 = *(const float4*)(arow + kb + 4);
      float4 a2 = *(const float4*)(arow + kb + 8);
      float4 a3 = *(const float4*)(arow + kb + 12);
      v16bf A;
      A[0] = (__bf16)(a0.x * s);  A[1] = (__bf16)(a0.y * s);
      A[2] = (__bf16)(a0.z * s);  A[3] = (__bf16)(a0.w * s);
      A[4] = (__bf16)(a1.x * s);  A[5] = (__bf16)(a1.y * s);
      A[6] = (__bf16)(a1.z * s);  A[7] = (__bf16)(a1.w * s);
      A[8] = (__bf16)(a2.x * s);  A[9] = (__bf16)(a2.y * s);
      A[10] = (__bf16)(a2.z * s); A[11] = (__bf16)(a2.w * s);
      A[12] = (__bf16)(a3.x * s); A[13] = (__bf16)(a3.y * s);
      A[14] = (__bf16)(a3.z * s); A[15] = (__bf16)(a3.w * s);
#pragma unroll
      for (int ct = 0; ct < 8; ++ct) {
        v16bf Bf = Bfrag[(size_t)(((rr * 4 + kt) * 8 + ct) * 32) + lane];
        acc[ct] = __builtin_amdgcn_wmma_f32_16x16x32_bf16(
            false, A, false, Bf, (short)0, acc[ct], false, false);
      }
    }
  };

  for (int r = 0; r < RGCN_R; ++r) {
    float c = cnt[(size_t)node * RGCN_R + r];
    float s = 1.0f / fmaxf(c, 1.0f);
    mac_operand(agg + ((size_t)node * RGCN_R + r) * RGCN_D, s, r);
  }
  mac_operand(x + (size_t)node * RGCN_D, 1.0f, RGCN_R);  // root term

  // Epilogue: C/D layout -> lanes 0-15: VGPR v = row v; lanes 16-31: row 8+v.
#pragma unroll
  for (int ct = 0; ct < 8; ++ct) {
#pragma unroll
    for (int v = 0; v < 8; ++v) {
      int nidx = tile0 + half * 8 + v;
      if (nidx < nNodes) {
        int col = ct * 16 + ncol;
        h[(size_t)nidx * RGCN_D + col] = fmaxf(acc[ct][v] + bias[col], 0.0f);
      }
    }
  }

  // Re-zero this wave's agg rows for the layer-2 scatter (sole reader).
  float4 z = make_float4(0.f, 0.f, 0.f, 0.f);
  int nrows = nNodes - tile0;
  nrows = nrows < 0 ? 0 : (nrows > 16 ? 16 : nrows);
  int span = nrows * RGCN_R * RGCN_D;
  float* base = agg + (size_t)tile0 * RGCN_R * RGCN_D;
  for (int i = lane * 4; i < span; i += 32 * 4) *(float4*)(base + i) = z;
}

// ----------------------------------------------------------- layer 2 (VALU)
// d_out = 2: far too thin for 16-wide WMMA tiles; one wave per node, lanes
// split K, shuffle-reduce the two output dots.
__global__ __launch_bounds__(256) void k_layer2(
    const float* __restrict__ agg, const float* __restrict__ cnt,
    const float* __restrict__ h, const float* __restrict__ Wrel2,
    const float* __restrict__ Wroot2, const float* __restrict__ b2,
    float* __restrict__ out, int nNodes) {
  int n = (blockIdx.x * blockDim.x + threadIdx.x) >> 5;
  int lane = threadIdx.x & 31;
  if (n >= nNodes) return;
  int k0 = lane * 4;
  float acc0 = 0.f, acc1 = 0.f;
  for (int r = 0; r < RGCN_R; ++r) {
    float c = cnt[(size_t)n * RGCN_R + r];
    float s = 1.0f / fmaxf(c, 1.0f);
    float4 a = *(const float4*)(agg + ((size_t)n * RGCN_R + r) * RGCN_D + k0);
    const float* w = Wrel2 + ((size_t)r * RGCN_D + k0) * 2;
    acc0 += s * (a.x * w[0] + a.y * w[2] + a.z * w[4] + a.w * w[6]);
    acc1 += s * (a.x * w[1] + a.y * w[3] + a.z * w[5] + a.w * w[7]);
  }
  float4 hx = *(const float4*)(h + (size_t)n * RGCN_D + k0);
  const float* wr = Wroot2 + (size_t)k0 * 2;
  acc0 += hx.x * wr[0] + hx.y * wr[2] + hx.z * wr[4] + hx.w * wr[6];
  acc1 += hx.x * wr[1] + hx.y * wr[3] + hx.z * wr[5] + hx.w * wr[7];
#pragma unroll
  for (int off = 16; off > 0; off >>= 1) {
    acc0 += __shfl_down(acc0, off, 32);
    acc1 += __shfl_down(acc1, off, 32);
  }
  if (lane == 0) {
    out[(size_t)n * 2 + 0] = acc0 + b2[0];
    out[(size_t)n * 2 + 1] = acc1 + b2[1];
  }
}

// ---------------------------------------------------------------------------
extern "C" void kernel_launch(void* const* d_in, const int* in_sizes, int n_in,
                              void* d_out, int out_size, void* d_ws, size_t ws_size,
                              hipStream_t stream) {
  const float* x      = (const float*)d_in[0];
  const int*   ei     = (const int*)d_in[1];   // [2,E] flat: src then dst
  const int*   et     = (const int*)d_in[2];   // [E]
  const float* Wrel1  = (const float*)d_in[3]; // [8,128,128]
  const float* Wroot1 = (const float*)d_in[4]; // [128,128]
  const float* b1     = (const float*)d_in[5]; // [128]
  const float* Wrel2  = (const float*)d_in[6]; // [8,128,2]
  const float* Wroot2 = (const float*)d_in[7]; // [128,2]
  const float* b2     = (const float*)d_in[8]; // [2]
  float* out = (float*)d_out;

  const int N = in_sizes[0] / RGCN_D;  // 50000
  const int E = in_sizes[2];           // 800000
  const int* src = ei;
  const int* dst = ei + E;

  // Workspace: agg [N*8*128] f32, cnt [N*8] f32, h [N*128] f32,
  //            Bpk [9*4*8*32*16] bf16 (prepacked weight fragments). ~232.3 MB
  float* agg = (float*)d_ws;
  float* cnt = agg + (size_t)N * RGCN_R * RGCN_D;
  float* h   = cnt + (size_t)N * RGCN_R;
  unsigned short* Bpk = (unsigned short*)(h + (size_t)N * RGCN_D);

  // 0) pre-swizzle layer-1 weights into bf16 WMMA B-fragments (once per call)
  k_prepack_w<<<(9 * 4 * 8 * 32 + 255) / 256, 256, 0, stream>>>(Wrel1, Wroot1,
                                                                Bpk);

  // 1) zero agg + cnt (contiguous)
  size_t nz = (size_t)N * RGCN_R * RGCN_D + (size_t)N * RGCN_R;
  k_zero_f4<<<8192, 256, 0, stream>>>(agg, nz);

  // 2) scatter x into per-(node,rel) sums + counts
  int sblocks = (E * 32 + 255) / 256;
  k_scatter<<<sblocks, 256, 0, stream>>>(x, src, dst, et, agg, cnt, E, 1);

  // 3) layer 1: WMMA GEMMs, bias+ReLU -> h ; re-zeros agg for layer 2
  k_layer1_wmma<<<(N + 63) / 64, 128, 0, stream>>>(agg, cnt, x, Bpk, b1, h, N);

  // 4) scatter h into per-(node,rel) sums (counts unchanged: same graph)
  k_scatter<<<sblocks, 256, 0, stream>>>(h, src, dst, et, agg, cnt, E, 0);

  // 5) layer 2: thin (d_out=2) projection, wave-per-node
  k_layer2<<<(N * 32 + 255) / 256, 256, 0, stream>>>(agg, cnt, h, Wrel2,
                                                     Wroot2, b2, out, N);
}
